// NodeModel_20839181320255
// MI455X (gfx1250) — matrix operand order repaired and checked by reading, not compiled, not run on hardware
//
#include <hip/hip_runtime.h>

// NodeModel (graph-net node block) for MI455X / gfx1250, wave32.
// Stage 1: per-edge 2-layer MLP (13->32->32) via v_wmma_f32_16x16x32_f16,
//          scatter-add (f32 atomics) into summed[node][32] + counts[node].
// Stage 2: per-node 2-layer MLP (103->32->7) via the same WMMA path.
// f16 WMMA inputs, f32 accumulation. Weights hoisted per persistent wave.

typedef __attribute__((ext_vector_type(16))) _Float16 v16h;
typedef __attribute__((ext_vector_type(8)))  _Float16 v8h;
typedef __attribute__((ext_vector_type(8)))  float    v8f;

#define WAVES 8          // 256 threads / 32
#define NEG_SLOPE 0.01f

__device__ __forceinline__ float leaky(float v) {
    return fmaxf(v, NEG_SLOPE * v);    // exact for slope < 1
}

// A-layout reload: two 16B-aligned LDS b128 loads (runs 16 halfs apart)
__device__ __forceinline__ v16h load_a_tile(const _Float16* p) {
    const v8h* q = (const v8h*)p;
    v8h a = q[0], b = q[2];
    return __builtin_shufflevector(a, b, 0, 1, 2, 3, 4, 5, 6, 7,
                                          8, 9, 10, 11, 12, 13, 14, 15);
}

__global__ __launch_bounds__(256) void zero_ws_kernel(float* __restrict__ p, long long n) {
    long long i = (long long)blockIdx.x * blockDim.x + threadIdx.x;
    long long stride = (long long)gridDim.x * blockDim.x;
    for (; i < n; i += stride) p[i] = 0.f;
}

// ---------------- Stage 1: edge MLP + scatter ----------------
__global__ __launch_bounds__(256) void edge_mlp_scatter(
    const float* __restrict__ x,        // [N][7]
    const int*   __restrict__ ei,       // [2][E] (row, col)
    const float* __restrict__ ea,       // [E][6]
    const float* __restrict__ W1, const float* __restrict__ b1,   // [13][32],[32]
    const float* __restrict__ W2, const float* __restrict__ b2,   // [32][32],[32]
    float* __restrict__ summed,         // [N][32]
    float* __restrict__ counts,         // [N]
    int E, int nTiles)
{
    __shared__ __align__(16) _Float16 lds[WAVES][16 * 32];
    const int lane = threadIdx.x & 31;
    const int wave = threadIdx.x >> 5;
    const int hi = lane >> 4;
    const int lo = lane & 15;
    const int wave0   = blockIdx.x * WAVES + wave;
    const int wstride = gridDim.x * WAVES;

    // ---- weights/bias in WMMA B / C layouts (hoisted, per persistent wave) ----
    v16h B1[2], B2[2];
    v8f  C1[2], C2[2];
    for (int t = 0; t < 2; ++t) {
        int n = t * 16 + lo;
        for (int e = 0; e < 16; ++e) {
            int k = hi * 8 + e + (e >= 8 ? 8 : 0);   // B layout: K = hi*8 + e (+8 for e>=8), N = lo
            B1[t][e] = (_Float16)((k < 13) ? W1[k * 32 + n] : 0.f);
            B2[t][e] = (_Float16)(W2[k * 32 + n]);
        }
        float bb1 = b1[n], bb2 = b2[n];
        for (int j = 0; j < 8; ++j) { C1[t][j] = bb1; C2[t][j] = bb2; }
    }
    _Float16* L = lds[wave];

    for (int tile = wave0; tile < nTiles; tile += wstride) {
        const int base = __builtin_amdgcn_readfirstlane(tile) * 16;  // wave-uniform
        const bool full = (base + 16 <= E);

        // ---- A1: 16 edges x (13 pad 32) features in A layout ----
        // lane half 0: elements 0..7 <-> K=0..7 (x feats 0..6, ea feat 0)
        // lane half 1: elements 0..7 <-> K=8..15 (ea feats 1..5, then zeros)
        v16h A1;
        #pragma unroll
        for (int e = 0; e < 16; ++e) A1[e] = (_Float16)0.f;
        if (full || (base + lo < E)) {
            if (hi == 0) {
                int row = ei[base + lo];
                #pragma unroll
                for (int k = 0; k < 7; ++k) A1[k] = (_Float16)x[row * 7 + k];
                A1[7] = (_Float16)ea[(base + lo) * 6];
            } else {
                #pragma unroll
                for (int k = 0; k < 5; ++k) A1[k] = (_Float16)ea[(base + lo) * 6 + 1 + k];
            }
        }

        // ---- layer 1 + LeakyReLU ----
        v8f D1[2];
        #pragma unroll
        for (int t = 0; t < 2; ++t)
            D1[t] = __builtin_amdgcn_wmma_f32_16x16x32_f16(
                        false, A1, false, B1[t], (short)0, C1[t], false, false);

        #pragma unroll
        for (int t = 0; t < 2; ++t)
            #pragma unroll
            for (int j = 0; j < 8; ++j)    // D element j -> (M = j + hi*8, N = lo)
                L[(j + hi * 8) * 32 + t * 16 + lo] = (_Float16)leaky(D1[t][j]);
        asm volatile("s_wait_dscnt 0" ::: "memory");   // wave-local LDS RAW fence

        // ---- A2: hidden[16][32] in A layout ----
        v16h A2 = load_a_tile(L + lo * 32 + hi * 8);

        // ---- layer 2 ----
        v8f D2[2];
        #pragma unroll
        for (int t = 0; t < 2; ++t)
            D2[t] = __builtin_amdgcn_wmma_f32_16x16x32_f16(
                        false, A2, false, B2[t], (short)0, C2[t], false, false);

        // ---- scatter-add ----
        if (full) {
            int colu[16];
            #pragma unroll
            for (int jj = 0; jj < 16; ++jj) colu[jj] = ei[E + base + jj];  // uniform
            if (hi == 0) atomicAdd(&counts[ei[E + base + lo]], 1.0f);
            #pragma unroll
            for (int j = 0; j < 8; ++j) {
                int dst = hi ? colu[8 + j] : colu[j];
                atomicAdd(&summed[dst * 32 + lo],      D2[0][j]);
                atomicAdd(&summed[dst * 32 + 16 + lo], D2[1][j]);
            }
        } else {
            if (hi == 0 && base + lo < E) atomicAdd(&counts[ei[E + base + lo]], 1.0f);
            for (int j = 0; j < 8; ++j) {
                int eid2 = base + hi * 8 + j;
                if (eid2 < E) {
                    int dst = ei[E + eid2];
                    atomicAdd(&summed[dst * 32 + lo],      D2[0][j]);
                    atomicAdd(&summed[dst * 32 + 16 + lo], D2[1][j]);
                }
            }
        }
    }
}

// ---------------- Stage 2: node MLP ----------------
__global__ __launch_bounds__(256) void node_mlp(
    const float* __restrict__ x,        // [N][7]
    const float* __restrict__ u,        // [G][64]
    const int*   __restrict__ batch,    // [N]
    const float* __restrict__ summed, const float* __restrict__ counts,
    const float* __restrict__ W3, const float* __restrict__ b3,   // [103][32],[32]
    const float* __restrict__ W4, const float* __restrict__ b4,   // [32][7],[7]
    float* __restrict__ out,            // [N][7]
    int N, int nTiles)
{
    __shared__ __align__(16) _Float16 feat[WAVES][16 * 128];
    const int lane = threadIdx.x & 31;
    const int wave = threadIdx.x >> 5;
    const int hi = lane >> 4;
    const int lo = lane & 15;
    const int wave0   = blockIdx.x * WAVES + wave;
    const int wstride = gridDim.x * WAVES;

    // ---- weights/bias (hoisted) ----
    v16h B3[2][4], B4;
    v8f  C3[2], C4;
    for (int t = 0; t < 2; ++t) {
        int n = t * 16 + lo;
        for (int c = 0; c < 4; ++c)
            for (int e = 0; e < 16; ++e) {
                int k = c * 32 + hi * 8 + e + (e >= 8 ? 8 : 0);
                B3[t][c][e] = (_Float16)((k < 103) ? W3[k * 32 + n] : 0.f);
            }
        float bb = b3[n];
        for (int j = 0; j < 8; ++j) C3[t][j] = bb;
    }
    for (int e = 0; e < 16; ++e) {
        int k = hi * 8 + e + (e >= 8 ? 8 : 0);
        B4[e] = (_Float16)((lo < 7) ? W4[k * 7 + lo] : 0.f);
    }
    {
        float bb = (lo < 7) ? b4[lo] : 0.f;
        for (int j = 0; j < 8; ++j) C4[j] = bb;
    }
    _Float16* F = feat[wave];

    for (int tile = wave0; tile < nTiles; tile += wstride) {
        const int base = __builtin_amdgcn_readfirstlane(tile) * 16;
        const bool full = (base + 16 <= N);
        const int node = base + lo;

        // ---- stage features [16 nodes][128 K] (103 real + 25 pad) ----
        if (full || node < N) {
            int bg = batch[node];
            if (hi == 0) {  // K 0..63
                float inv = 1.f / fmaxf(counts[node], 1.f);
                #pragma unroll
                for (int k = 0; k < 7;  ++k) F[lo * 128 + k]      = (_Float16)x[node * 7 + k];
                #pragma unroll
                for (int k = 0; k < 32; ++k) F[lo * 128 + 7 + k]  = (_Float16)(summed[node * 32 + k] * inv);
                #pragma unroll
                for (int k = 0; k < 25; ++k) F[lo * 128 + 39 + k] = (_Float16)u[bg * 64 + k];
            } else {        // K 64..127
                #pragma unroll
                for (int k = 25; k < 64; ++k)   F[lo * 128 + 39 + k] = (_Float16)u[bg * 64 + k];
                #pragma unroll
                for (int k = 103; k < 128; ++k) F[lo * 128 + k] = (_Float16)0.f;
            }
        } else {
            // pad rows of a partial tile with zeros
            for (int k = 0; k < 64; ++k) F[lo * 128 + hi * 64 + k] = (_Float16)0.f;
        }
        asm volatile("s_wait_dscnt 0" ::: "memory");

        // ---- layer 3: K = 128 in 4 WMMA chunks per N-tile ----
        v8f acc[2];
        acc[0] = C3[0]; acc[1] = C3[1];
        #pragma unroll
        for (int c = 0; c < 4; ++c) {
            v16h A = load_a_tile(F + lo * 128 + c * 32 + hi * 8);
            #pragma unroll
            for (int t = 0; t < 2; ++t)
                acc[t] = __builtin_amdgcn_wmma_f32_16x16x32_f16(
                             false, A, false, B3[t][c], (short)0, acc[t], false, false);
        }

        // ---- LeakyReLU, restage hidden[16][32], layer 4 ----
        #pragma unroll
        for (int t = 0; t < 2; ++t)
            #pragma unroll
            for (int j = 0; j < 8; ++j)
                F[(j + hi * 8) * 32 + t * 16 + lo] = (_Float16)leaky(acc[t][j]);
        asm volatile("s_wait_dscnt 0" ::: "memory");

        v16h A4 = load_a_tile(F + lo * 32 + hi * 8);
        v8f D = __builtin_amdgcn_wmma_f32_16x16x32_f16(
                    false, A4, false, B4, (short)0, C4, false, false);

        if (full) {
            if (lo < 7) {
                #pragma unroll
                for (int j = 0; j < 8; ++j)
                    out[(base + hi * 8 + j) * 7 + lo] = D[j];
            }
        } else {
            for (int j = 0; j < 8; ++j) {
                int nd = base + hi * 8 + j;
                if (nd < N && lo < 7) out[nd * 7 + lo] = D[j];
            }
        }
    }
}

extern "C" void kernel_launch(void* const* d_in, const int* in_sizes, int n_in,
                              void* d_out, int out_size, void* d_ws, size_t ws_size,
                              hipStream_t stream) {
    const float* x     = (const float*)d_in[0];
    const int*   ei    = (const int*)  d_in[1];
    const float* ea    = (const float*)d_in[2];
    const float* u     = (const float*)d_in[3];
    const int*   batch = (const int*)  d_in[4];
    const float* W1 = (const float*)d_in[5],  *b1 = (const float*)d_in[6];
    const float* W2 = (const float*)d_in[7],  *b2 = (const float*)d_in[8];
    const float* W3 = (const float*)d_in[9],  *b3 = (const float*)d_in[10];
    const float* W4 = (const float*)d_in[11], *b4 = (const float*)d_in[12];
    float* out = (float*)d_out;

    const int N = in_sizes[0] / 7;   // 100000
    const int E = in_sizes[2] / 6;   // 3200000

    float* summed = (float*)d_ws;            // [N][32]
    float* counts = summed + (size_t)N * 32; // [N]

    // zero accumulators (graph-capture-safe, deterministic each call)
    long long zn = (long long)N * 33;
    zero_ws_kernel<<<1024, 256, 0, stream>>>(summed, zn);

    int eTiles  = (E + 15) / 16;                 // 200000
    int eBlocks = (eTiles + WAVES - 1) / WAVES;
    if (eBlocks > 1024) eBlocks = 1024;          // persistent waves, ~24 tiles each
    edge_mlp_scatter<<<eBlocks, 256, 0, stream>>>(x, ei, ea, W1, b1, W2, b2,
                                                  summed, counts, E, eTiles);

    int nTiles  = (N + 15) / 16;                 // 6250
    int nBlocks = (nTiles + WAVES - 1) / WAVES;
    if (nBlocks > 384) nBlocks = 384;
    node_mlp<<<nBlocks, 256, 0, stream>>>(x, u, batch, summed, counts,
                                          W3, b3, W4, b4, out, N, nTiles);
}
